// DarwinianRouter_43430709297941
// MI455X (gfx1250) — matrix operand order repaired and checked by reading, not compiled, not run on hardware
//
#include <hip/hip_runtime.h>
#include <hip/hip_bf16.h>

#define T_TOKENS 16384
#define E_EXPERTS 64
#define D_DIM 4096
#define TOPK 8

typedef __attribute__((ext_vector_type(16))) __bf16 v16bf;
typedef __attribute__((ext_vector_type(8)))  float  v8f;

struct Bfrag { v16bf hi; v16bf lo; };

// ---------------------------------------------------------------------------
// Kernel 1: L2-normalize each signature row, split into bf16 hi/lo pair.
// sig_hi/sig_lo laid out row-major [expert][k] (matches WMMA B fetch pattern).
// ---------------------------------------------------------------------------
__global__ __launch_bounds__(256) void sig_prep_kernel(
    const float* __restrict__ sig,
    __bf16* __restrict__ sig_hi,
    __bf16* __restrict__ sig_lo) {
  __shared__ float red[256];
  const int e = blockIdx.x;
  const float* row = sig + (size_t)e * D_DIM;

  float ss = 0.f;
  for (int k = threadIdx.x; k < D_DIM; k += 256) {
    float v = row[k];
    ss += v * v;
  }
  red[threadIdx.x] = ss;
  __syncthreads();
  for (int s = 128; s > 0; s >>= 1) {
    if (threadIdx.x < s) red[threadIdx.x] += red[threadIdx.x + s];
    __syncthreads();
  }
  const float inv = 1.0f / fmaxf(sqrtf(red[0]), 1e-12f);

  for (int k = threadIdx.x; k < D_DIM; k += 256) {
    float v = row[k] * inv;
    __bf16 h = (__bf16)v;
    sig_hi[(size_t)e * D_DIM + k] = h;
    sig_lo[(size_t)e * D_DIM + k] = (__bf16)(v - (float)h);
  }
}

// ---------------------------------------------------------------------------
// Kernel 2: fused  (x/||x||) @ sig_norm^T  ->  top-8 (relu vals + indices).
// One wave32 per 16-token tile; 4 expert tiles of 16; K streamed in chunks
// of 32 with bf16 hi/lo split (3x v_wmma_f32_16x16x32_bf16 per tile pair).
// Software-pipelined: A (x) prefetched one K-chunk ahead, B one N-tile ahead.
// ---------------------------------------------------------------------------
__global__ __launch_bounds__(256) void router_wmma_kernel(
    const float*  __restrict__ x,
    const __bf16* __restrict__ sig_hi,
    const __bf16* __restrict__ sig_lo,
    float* __restrict__ w_out,
    int*   __restrict__ i_out) {
  __shared__ float scores[8][16][E_EXPERTS];   // 32 KB

  const int wave = threadIdx.x >> 5;
  const int lane = threadIdx.x & 31;
  const int col  = lane & 15;     // A: token row loaded; B: expert col; C: N
  const int hlf  = lane >> 4;     // lane half selects K sub-range / M half
  const int kOffA = hlf * 16;     // A frag: K = kOffA..+15 (contiguous)
  const int kOffB = hlf * 16;     // B frag: K = kOffB..+15 (contiguous)

  const int tile = blockIdx.x * 8 + wave;       // 16-token tile id
  const float* xrow = x + (size_t)(tile * 16 + col) * D_DIM;

  v8f acc[4];
#pragma unroll
  for (int nt = 0; nt < 4; ++nt)
#pragma unroll
    for (int j = 0; j < 8; ++j) acc[nt][j] = 0.f;

  float ssq = 0.f;

  // ---- software pipeline: prime A chunk 0 and B fragment (k=0, nt=0) ----
  float4 fa[2][4];
#pragma unroll
  for (int g = 0; g < 2; ++g) {
    fa[0][2 * g]     = *(const float4*)(xrow + kOffA + 8 * g);
    fa[0][2 * g + 1] = *(const float4*)(xrow + kOffA + 8 * g + 4);
  }
  Bfrag bcur;
  {
    const size_t boff = (size_t)col * D_DIM + kOffB;
    bcur.hi = *(const v16bf*)(sig_hi + boff);
    bcur.lo = *(const v16bf*)(sig_lo + boff);
  }

#pragma unroll 2
  for (int k = 0; k < D_DIM; k += 32) {
    const int buf = (k >> 5) & 1;

    // ---- prefetch next A chunk (x stream: the HBM-critical loads) ----
    if (k + 32 < D_DIM) {
      const float* xn = xrow + k + 32 + kOffA;
#pragma unroll
      for (int g = 0; g < 2; ++g) {
        fa[buf ^ 1][2 * g]     = *(const float4*)(xn + 8 * g);
        fa[buf ^ 1][2 * g + 1] = *(const float4*)(xn + 8 * g + 4);
      }
    }

    // ---- convert current A to bf16 hi/lo, accumulate sum of squares ----
    float f[16];
#pragma unroll
    for (int g = 0; g < 4; ++g) {
      f[4 * g + 0] = fa[buf][g].x; f[4 * g + 1] = fa[buf][g].y;
      f[4 * g + 2] = fa[buf][g].z; f[4 * g + 3] = fa[buf][g].w;
    }
    v16bf a_hi, a_lo;
#pragma unroll
    for (int i = 0; i < 16; ++i) {
      ssq += f[i] * f[i];
      __bf16 h = (__bf16)f[i];
      a_hi[i] = h;
      a_lo[i] = (__bf16)(f[i] - (float)h);
    }

    // ---- 4 expert tiles, B pipelined one tile ahead ----
#pragma unroll
    for (int nt = 0; nt < 4; ++nt) {
      // Prefetch next B fragment (nt+1, or nt=0 of next k-chunk).
      // Tail overshoot (k=D_DIM, nt=0) stays inside the sig arrays
      // (lands in rows col+1 <= 16 of 64), so no guard branch needed.
      const int nnt = (nt + 1) & 3;
      const int nk  = (nt == 3) ? (k + 32) : k;
      const size_t boff = (size_t)(nnt * 16 + col) * D_DIM + nk + kOffB;
      Bfrag bnext;
      bnext.hi = *(const v16bf*)(sig_hi + boff);
      bnext.lo = *(const v16bf*)(sig_lo + boff);

      acc[nt] = __builtin_amdgcn_wmma_f32_16x16x32_bf16(
          false, a_hi, false, bcur.hi, (short)0, acc[nt], false, false);
      acc[nt] = __builtin_amdgcn_wmma_f32_16x16x32_bf16(
          false, a_lo, false, bcur.hi, (short)0, acc[nt], false, false);
      acc[nt] = __builtin_amdgcn_wmma_f32_16x16x32_bf16(
          false, a_hi, false, bcur.lo, (short)0, acc[nt], false, false);

      bcur = bnext;
    }
  }

  // ---- per-token 1/||x||: lanes r and r+16 hold complementary K partials ----
  ssq += __shfl_xor(ssq, 16, 32);
  const float invn = 1.0f / fmaxf(sqrtf(ssq), 1e-12f);  // token = lane&15

  // ---- scale + scatter C (lane holds N=col, M=j+8*hlf) into LDS ----
#pragma unroll
  for (int nt = 0; nt < 4; ++nt) {
#pragma unroll
    for (int j = 0; j < 8; ++j) {
      const int m = j + 8 * hlf;
      scores[wave][m][nt * 16 + col] = acc[nt][j] * __shfl(invn, m, 32);
    }
  }
  __syncthreads();

  // ---- top-8 per token: lanes 0-15 each own one token of this wave's tile ----
  if (lane < 16) {
    const float* s = scores[wave][lane];
    float bv[TOPK];
    int   bi[TOPK];
#pragma unroll
    for (int j = 0; j < TOPK; ++j) { bv[j] = -3.402823466e38f; bi[j] = 0; }
    for (int e = 0; e < E_EXPERTS; ++e) {
      float v = s[e];
      if (v > bv[TOPK - 1]) {            // strict '>' keeps lowest index on ties
        int p = TOPK - 1;
        while (p > 0 && v > bv[p - 1]) {
          bv[p] = bv[p - 1]; bi[p] = bi[p - 1]; --p;
        }
        bv[p] = v; bi[p] = e;
      }
    }
    const int trow = tile * 16 + lane;
#pragma unroll
    for (int j = 0; j < TOPK; ++j) {
      w_out[trow * TOPK + j] = fmaxf(bv[j], 0.0f);   // relu(top_vals)
      i_out[trow * TOPK + j] = bi[j];
    }
  }
}

// ---------------------------------------------------------------------------
extern "C" void kernel_launch(void* const* d_in, const int* in_sizes, int n_in,
                              void* d_out, int out_size, void* d_ws, size_t ws_size,
                              hipStream_t stream) {
  const float* x   = (const float*)d_in[0];
  const float* sig = (const float*)d_in[1];
  // d_in[2] = top_k scalar (== 8, compile-time here)

  __bf16* sig_hi = (__bf16*)d_ws;                               // 512 KB
  __bf16* sig_lo = sig_hi + (size_t)E_EXPERTS * D_DIM;          // 512 KB

  float* w_out = (float*)d_out;                                 // [16384, 8] f32
  int*   i_out = (int*)(w_out + (size_t)T_TOKENS * TOPK);       // [16384, 8] i32

  sig_prep_kernel<<<E_EXPERTS, 256, 0, stream>>>(sig, sig_hi, sig_lo);
  router_wmma_kernel<<<T_TOKENS / 128, 256, 0, stream>>>(x, sig_hi, sig_lo,
                                                         w_out, i_out);
}
